// MultiAttentionGRU_50783693308404
// MI455X (gfx1250) — matrix operand order repaired and checked by reading, not compiled, not run on hardware
//
#include <hip/hip_runtime.h>
#include <math.h>

// ---------------- problem constants ----------------
#define HID      64
#define HEADS    5
#define MAXDEG   4
#define VOCAB    5000
#define NCLASS   4
#define N_LEAVES 24576
#define N_PARENTS 8192
#define N_NODES  (N_LEAVES + N_PARENTS)
#define WORDS    32
#define NEGV     (-1.0e9f)

typedef __attribute__((ext_vector_type(16))) _Float16 v16h;
typedef __attribute__((ext_vector_type(8)))  float    v8f;

// =====================================================================
// WMMA GEMM: C[M,N] = A[M,K] * B[K,N]  (row-major f32 in/out, f16
// operands, f32 accumulate).  One wave computes a 64x16 strip: the B
// fragment is loaded once per K-step and reused by 4 WMMAs (register-
// level B reuse).  M multiple of 64, N multiple of 16, K multiple of 32.
// Fragment layouts per CDNA5 ISA 7.12.2 (wave32); EXEC all-ones.
// =====================================================================
__device__ __forceinline__ v16h load_a_frag(const float* __restrict__ A,
                                            int row, int K, int kb, bool hi) {
  v16h a;
  const float* arow = A + (size_t)row * K + kb + (hi ? 8 : 0);
#pragma unroll
  for (int e = 0; e < 8; ++e) {
    a[e]     = (_Float16)arow[e];        // K = base + e
    a[e + 8] = (_Float16)arow[e + 16];   // K = base + 16 + e
  }
  return a;
}

__global__ void gemm_wmma_f16(const float* __restrict__ A,
                              const float* __restrict__ B,
                              float* __restrict__ C,
                              int M, int N, int K) {
  const int lane = threadIdx.x;          // 0..31, one wave
  const int m0 = blockIdx.x * 64;        // 64-row strip
  const int n0 = blockIdx.y * 16;
  const int rsel = lane & 15;
  const bool hi = lane >= 16;

  v8f c0 = {}, c1 = {}, c2 = {}, c3 = {};
  for (int kb = 0; kb < K; kb += 32) {
    // ---- B fragment (shared by 4 WMMAs): lane<16 -> K=kb+0..7,16..23, lane>=16 -> K=kb+8..15,24..31
    v16h b;
    const float* bcol = B + (size_t)(kb + (hi ? 8 : 0)) * N + n0 + rsel;
#pragma unroll
    for (int e = 0; e < 8; ++e) { b[e] = (_Float16)bcol[(size_t)e * N]; b[e + 8] = (_Float16)bcol[(size_t)(e + 16) * N]; }

    v16h a0 = load_a_frag(A, m0 + rsel,      K, kb, hi);
    v16h a1 = load_a_frag(A, m0 + 16 + rsel, K, kb, hi);
    v16h a2 = load_a_frag(A, m0 + 32 + rsel, K, kb, hi);
    v16h a3 = load_a_frag(A, m0 + 48 + rsel, K, kb, hi);

    c0 = __builtin_amdgcn_wmma_f32_16x16x32_f16(false, a0, false, b, (short)0, c0, false, false);
    c1 = __builtin_amdgcn_wmma_f32_16x16x32_f16(false, a1, false, b, (short)0, c1, false, false);
    c2 = __builtin_amdgcn_wmma_f32_16x16x32_f16(false, a2, false, b, (short)0, c2, false, false);
    c3 = __builtin_amdgcn_wmma_f32_16x16x32_f16(false, a3, false, b, (short)0, c3, false, false);
  }
  // ---- store: lane holds col n0+rsel, rows (tile base) + (hi?8:0) + v
  {
    float* cr = C + (size_t)(m0 + (hi ? 8 : 0)) * N + n0 + rsel;
#pragma unroll
    for (int v = 0; v < 8; ++v) cr[(size_t)v * N] = c0[v];
  }
  {
    float* cr = C + (size_t)(m0 + 16 + (hi ? 8 : 0)) * N + n0 + rsel;
#pragma unroll
    for (int v = 0; v < 8; ++v) cr[(size_t)v * N] = c1[v];
  }
  {
    float* cr = C + (size_t)(m0 + 32 + (hi ? 8 : 0)) * N + n0 + rsel;
#pragma unroll
    for (int v = 0; v < 8; ++v) cr[(size_t)v * N] = c2[v];
  }
  {
    float* cr = C + (size_t)(m0 + 48 + (hi ? 8 : 0)) * N + n0 + rsel;
#pragma unroll
    for (int v = 0; v < 8; ++v) cr[(size_t)v * N] = c3[v];
  }
}

// =====================================================================
// Small prep kernels (weight re-layouts so every GEMM is A*B row-major)
// =====================================================================
__global__ void transpose_E(const float* __restrict__ E, float* __restrict__ Et) {
  int id = blockIdx.x * blockDim.x + threadIdx.x;      // Et[v*64+d] = E[d*5000+v]
  if (id >= VOCAB * HID) return;
  int v = id >> 6, d = id & 63;
  Et[id] = E[(size_t)d * VOCAB + v];
}

__global__ void pack_wcat3(const float* __restrict__ Wz, const float* __restrict__ Wr,
                           const float* __restrict__ Wh, float* __restrict__ Bc) {
  int id = blockIdx.x * blockDim.x + threadIdx.x;      // Bc[j*192 + g*64+i] = Wg[i*64+j]
  if (id >= 64 * 192) return;
  int j = id / 192, col = id % 192, g = col >> 6, i = col & 63;
  const float* W = (g == 0) ? Wz : ((g == 1) ? Wr : Wh);
  Bc[id] = W[i * 64 + j];
}

__global__ void pack_uhB(const float* __restrict__ Uh, float* __restrict__ B) {
  int id = blockIdx.x * blockDim.x + threadIdx.x;      // B[j*64+i] = Uh[i*64+j]
  if (id >= 4096) return;
  int j = id >> 6, i = id & 63;
  B[id] = Uh[i * 64 + j];
}

__global__ void pack_wqcat(const float* __restrict__ WQ, float* __restrict__ B) {
  int id = blockIdx.x * blockDim.x + threadIdx.x;      // B[j*320 + m*64+i] = WQ[m,i,j]
  if (id >= 64 * 320) return;
  int j = id / 320, col = id % 320, m = col >> 6, i = col & 63;
  B[id] = WQ[m * 4096 + i * 64 + j];
}

__global__ void pack_wkbd(const float* __restrict__ WK, float* __restrict__ B) {
  // qw = q @ B ;  B[(m*64+i)*320 + m2*64+j] = (m==m2) ? WK[m,j,i] : 0
  int id = blockIdx.x * blockDim.x + threadIdx.x;
  if (id >= 320 * 320) return;
  int k = id / 320, col = id % 320;
  int m = k >> 6, i = k & 63, m2 = col >> 6, j = col & 63;
  B[id] = (m == m2) ? WK[m * 4096 + j * 64 + i] : 0.0f;
}

__global__ void pack_wot(const float* __restrict__ WO, float* __restrict__ B) {
  int id = blockIdx.x * blockDim.x + threadIdx.x;      // B[k*64+i] = WO[i*320+k]
  if (id >= 320 * 64) return;
  int k = id >> 6, i = id & 63;
  B[id] = WO[i * 320 + k];
}

__global__ void prep_const(const float* __restrict__ Uz, const float* __restrict__ Ur,
                           const float* __restrict__ lnb,
                           float* __restrict__ uzc, float* __restrict__ urc) {
  int i = threadIdx.x;                                  // 64 threads
  float a = 0.f, b = 0.f;
  for (int j = 0; j < 64; ++j) {
    a += Uz[i * 64 + j] * lnb[j];
    b += Ur[i * 64 + j] * lnb[j];
  }
  uzc[i] = a; urc[i] = b;
}

// =====================================================================
// xe[n,d] = sum_l x_word[n,l] * E_t[x_index[n,l], d]   (E_t L2-resident)
// =====================================================================
__global__ void xe_kernel(const float* __restrict__ xw, const int* __restrict__ xi,
                          const float* __restrict__ Et, float* __restrict__ xe) {
  int n = blockIdx.x, d = threadIdx.x;                  // 64 threads / node
  const int*   idx = xi + (size_t)n * WORDS;
  const float* w   = xw + (size_t)n * WORDS;
  float acc = 0.f;
#pragma unroll 4
  for (int l = 0; l < WORDS; ++l) acc += w[l] * Et[(size_t)idx[l] * HID + d];
  xe[(size_t)n * HID + d] = acc;
}

// =====================================================================
// Leaf elementwise: z,r from gate pre-activations; t = ln_b * r
// =====================================================================
__global__ void leaf_zr(const float* __restrict__ G, const float* __restrict__ uzc,
                        const float* __restrict__ urc, const float* __restrict__ bz,
                        const float* __restrict__ br, const float* __restrict__ lnb,
                        float* __restrict__ zst, float* __restrict__ t) {
  int id = blockIdx.x * blockDim.x + threadIdx.x;
  if (id >= N_LEAVES * HID) return;
  int n = id >> 6, i = id & 63;
  float z = 1.f / (1.f + expf(-(G[(size_t)n * 192 + i] + uzc[i] + bz[i])));
  float r = 1.f / (1.f + expf(-(G[(size_t)n * 192 + 64 + i] + urc[i] + br[i])));
  zst[id] = z;
  t[id]   = lnb[i] * r;
}

__global__ void leaf_final(const float* __restrict__ G, const float* __restrict__ O2,
                           const float* __restrict__ zst, const float* __restrict__ bh,
                           const float* __restrict__ lnb, float* __restrict__ leaf_h) {
  int id = blockIdx.x * blockDim.x + threadIdx.x;
  if (id >= N_LEAVES * HID) return;
  int n = id >> 6, i = id & 63;
  float c = tanhf(G[(size_t)n * 192 + 128 + i] + bh[i] + O2[id]);
  float z = zst[id];
  leaf_h[id] = z * lnb[i] + (1.f - z) * c;
}

// =====================================================================
// Attention: logits -> softmax -> att (all 20 per parent) and
// pre320[p, m*64+i] = sum_{leaf children c} att[m,c]*leaf_h[ch_c, i]
// =====================================================================
__global__ void att_kernel(const float* __restrict__ qw, const float* __restrict__ xe,
                           const int* __restrict__ tree, const float* __restrict__ leaf_h,
                           float* __restrict__ att, float* __restrict__ pre320) {
  int p = blockIdx.x, tid = threadIdx.x;                // 64 threads / parent
  __shared__ float lg[HEADS * MAXDEG];
  __shared__ float at[HEADS * MAXDEG];
  __shared__ int   ch[MAXDEG];
  if (tid < MAXDEG) ch[tid] = tree[p * (MAXDEG + 1) + tid];
  __syncthreads();
  if (tid < HEADS * MAXDEG) {
    int m = tid / MAXDEG, c = tid % MAXDEG;
    int ci = ch[c];
    float v = NEGV;
    if (ci > -1) {
      const float* qq = qw + (size_t)p * 320 + m * 64;
      const float* x  = xe + (size_t)ci * HID;
      float a = 0.f;
      for (int j = 0; j < HID; ++j) a += qq[j] * x[j];
      v = a * ((m == 0) ? 0.125f : 8.0f);               // scale = [1/sqrt(64), sqrt(64)...]
    }
    lg[tid] = v;
  }
  __syncthreads();
  if (tid < HEADS) {                                    // softmax over 4 children
    float mx = lg[tid * 4];
    for (int c = 1; c < 4; ++c) mx = fmaxf(mx, lg[tid * 4 + c]);
    float e[4], s = 0.f;
    for (int c = 0; c < 4; ++c) { e[c] = expf(lg[tid * 4 + c] - mx); s += e[c]; }
    for (int c = 0; c < 4; ++c) {
      float a = e[c] / s;
      at[tid * 4 + c] = a;
      att[(size_t)p * 20 + tid * 4 + c] = a;
    }
  }
  __syncthreads();
  for (int u = tid; u < HEADS * HID; u += 64) {
    int m = u >> 6, i = u & 63;
    float a = 0.f;
    for (int c = 0; c < MAXDEG; ++c) {
      int ci = ch[c];
      if (ci > -1 && ci < N_LEAVES) a += at[m * 4 + c] * leaf_h[(size_t)ci * HID + i];
    }
    pre320[(size_t)p * 320 + u] = a;
  }
}

// =====================================================================
// Sequential scan: single persistent workgroup (256 threads, 8 waves).
// LDS-resident WO^T (80KB) + Uz^T/Ur^T/Uh^T (48KB) -> per-step latency
// minimized; only ~1KB of precomputed per-parent data streams from L2.
// =====================================================================
__global__ void __launch_bounds__(256)
seq_kernel(const float* __restrict__ WOT,   // [320*64]  WO^T
           const float* __restrict__ Uz, const float* __restrict__ Ur,
           const float* __restrict__ Uh,
           const float* __restrict__ gP,    // [8192*192] W*pxe per parent
           const float* __restrict__ vp,    // [8192*64]
           const float* __restrict__ att,   // [8192*20]
           const float* __restrict__ bz, const float* __restrict__ br,
           const float* __restrict__ bh,
           const float* __restrict__ lng, const float* __restrict__ lnb,
           float* __restrict__ root) {
  extern __shared__ float sm[];
  float* WOt  = sm;               // 20480  : WOt[t*64+i] = WO[i,t]
  float* UzT  = sm + 20480;       // 4096   : UzT[j*64+i] = Uz[i,j]
  float* UrT  = sm + 24576;       // 4096
  float* UhT  = sm + 28672;       // 4096
  float* s    = sm + 32768;       // 320
  float* part = sm + 33088;       // 256
  float* hraw = sm + 33344;       // 64
  float* htl  = sm + 33408;       // 64
  float* hbuf = sm + 33472;       // 64
  float* zb   = sm + 33536;       // 64
  float* rb   = sm + 33600;       // 64
  float* red  = sm + 33664;       // 32
  float* redq = sm + 33696;       // 32
  float* gL   = sm + 33728;       // 64 ln_g
  float* bL   = sm + 33792;       // 64 ln_b
  float* bzL  = sm + 33856;       // 64
  float* brL  = sm + 33920;       // 64
  float* bhL  = sm + 33984;       // 64
  float* scal = sm + 34048;       // 2 (mu, inv_std)

  const int tid = threadIdx.x;

  for (int u = tid; u < 20480; u += 256) WOt[u] = WOT[u];
  for (int u = tid; u < 4096; u += 256) {
    int j = u >> 6, i = u & 63;
    UzT[u] = Uz[i * 64 + j];
    UrT[u] = Ur[i * 64 + j];
    UhT[u] = Uh[i * 64 + j];
  }
  if (tid < 64) {
    gL[tid] = lng[tid]; bL[tid] = lnb[tid];
    bzL[tid] = bz[tid]; brL[tid] = br[tid]; bhL[tid] = bh[tid];
    hbuf[tid] = 0.f;
  }
  __syncthreads();

  for (int p = 0; p < N_PARENTS; ++p) {
    // s[m*64+j] = att[p,m,3] * h_prev[j]   (att3 ~ 0 for p==0 via NEG mask)
    for (int t = tid; t < 320; t += 256)
      s[t] = att[(size_t)p * 20 + (t >> 6) * 4 + 3] * hbuf[t & 63];
    __syncthreads();

    // u = WO @ s   (64 rows x 4 K-slices of 80)
    {
      int slice = tid >> 6, i = tid & 63;
      float acc = 0.f;
      int t0 = slice * 80;
#pragma unroll 8
      for (int c = 0; c < 80; ++c) acc += WOt[(t0 + c) * 64 + i] * s[t0 + c];
      part[tid] = acc;
    }
    __syncthreads();
    if (tid < 64)
      hraw[tid] = part[tid] + part[64 + tid] + part[128 + tid] + part[192 + tid]
                + vp[(size_t)p * 64 + tid];
    __syncthreads();

    // LayerNorm over 64
    if (tid < 32) {
      float a = hraw[tid], b = hraw[tid + 32];
      red[tid] = a + b;
      redq[tid] = a * a + b * b;
    }
    __syncthreads();
    if (tid == 0) {
      float sumv = 0.f, sq = 0.f;
      for (int k = 0; k < 32; ++k) { sumv += red[k]; sq += redq[k]; }
      float mu = sumv * (1.f / 64.f);
      float var = sq * (1.f / 64.f) - mu * mu;
      scal[0] = mu;
      scal[1] = rsqrtf(var + 1e-5f);
    }
    __syncthreads();
    if (tid < 64)
      htl[tid] = (hraw[tid] - scal[0]) * scal[1] * gL[tid] + bL[tid];
    __syncthreads();

    // z, r  (128 threads, each a 64-MAC matvec row)
    if (tid < 128) {
      int g = tid >> 6, i = tid & 63;
      const float* UT = g ? UrT : UzT;
      float acc = gP[(size_t)p * 192 + g * 64 + i] + (g ? brL[i] : bzL[i]);
#pragma unroll 8
      for (int j = 0; j < 64; ++j) acc += UT[j * 64 + i] * htl[j];
      float v = 1.f / (1.f + expf(-acc));
      if (g) rb[i] = v; else zb[i] = v;
    }
    __syncthreads();
    if (tid < 64) s[tid] = htl[tid] * rb[tid];   // reuse s as (h_tilde * r)
    __syncthreads();
    if (tid < 64) {
      int i = tid;
      float acc = gP[(size_t)p * 192 + 128 + i] + bhL[i];
#pragma unroll 8
      for (int j = 0; j < 64; ++j) acc += UhT[j * 64 + i] * s[j];
      float c = tanhf(acc);
      hbuf[i] = zb[i] * htl[i] + (1.f - zb[i]) * c;
    }
    __syncthreads();
  }

  if (tid < 64) root[tid] = hbuf[tid];
}

// =====================================================================
// Final: loss = sum((y - softmax(W_out@root + b_out))^2)
// =====================================================================
__global__ void loss_kernel(const float* __restrict__ root, const float* __restrict__ Wout,
                            const float* __restrict__ bout, const float* __restrict__ y,
                            float* __restrict__ out) {
  if (threadIdx.x == 0) {
    float l[NCLASS], mx = -1e30f;
    for (int k = 0; k < NCLASS; ++k) {
      float a = bout[k];
      for (int i = 0; i < HID; ++i) a += Wout[k * HID + i] * root[i];
      l[k] = a; mx = fmaxf(mx, a);
    }
    float e[NCLASS], su = 0.f;
    for (int k = 0; k < NCLASS; ++k) { e[k] = expf(l[k] - mx); su += e[k]; }
    float loss = 0.f;
    for (int k = 0; k < NCLASS; ++k) {
      float d = y[k] - e[k] / su;
      loss += d * d;
    }
    out[0] = loss;
  }
}

// =====================================================================
// Host orchestration
// =====================================================================
extern "C" void kernel_launch(void* const* d_in, const int* in_sizes, int n_in,
                              void* d_out, int out_size, void* d_ws, size_t ws_size,
                              hipStream_t stream) {
  const float* x_word = (const float*)d_in[0];
  const int*   x_index = (const int*)d_in[1];
  const int*   tree = (const int*)d_in[2];
  const float* y = (const float*)d_in[3];
  const float* E_bu = (const float*)d_in[4];
  const float* W_z = (const float*)d_in[5];
  const float* U_z = (const float*)d_in[6];
  const float* b_z = (const float*)d_in[7];
  const float* W_r = (const float*)d_in[8];
  const float* U_r = (const float*)d_in[9];
  const float* b_r = (const float*)d_in[10];
  const float* W_h = (const float*)d_in[11];
  const float* U_h = (const float*)d_in[12];
  const float* b_h = (const float*)d_in[13];
  const float* W_out = (const float*)d_in[14];
  const float* b_out = (const float*)d_in[15];
  const float* WQ = (const float*)d_in[16];
  const float* WK = (const float*)d_in[17];
  const float* WO = (const float*)d_in[18];
  const float* ln_g = (const float*)d_in[19];
  const float* ln_b = (const float*)d_in[20];
  float* out = (float*)d_out;

  // ---- workspace layout (floats) ----
  float* ws = (float*)d_ws;
  size_t o = 0;
  float* ET    = ws + o; o += (size_t)VOCAB * HID;       // 320000
  float* XE    = ws + o; o += (size_t)N_NODES * HID;     // 2097152
  float* LEAFH = ws + o; o += (size_t)N_LEAVES * HID;
  float* G     = ws + o; o += (size_t)N_LEAVES * 192;
  float* ZST   = ws + o; o += (size_t)N_LEAVES * HID;
  float* T     = ws + o; o += (size_t)N_LEAVES * HID;
  float* O2    = ws + o; o += (size_t)N_LEAVES * HID;
  float* Q     = ws + o; o += (size_t)N_PARENTS * 320;
  float* QW    = ws + o; o += (size_t)N_PARENTS * 320;
  float* GP    = ws + o; o += (size_t)N_PARENTS * 192;
  float* ATT   = ws + o; o += (size_t)N_PARENTS * 20;
  float* PRE   = ws + o; o += (size_t)N_PARENTS * 320;
  float* VP    = ws + o; o += (size_t)N_PARENTS * HID;
  float* BCAT  = ws + o; o += 64 * 192;
  float* BUH   = ws + o; o += 64 * 64;
  float* BWQ   = ws + o; o += 64 * 320;
  float* BKBD  = ws + o; o += 320 * 320;
  float* BWOT  = ws + o; o += 320 * 64;
  float* UZC   = ws + o; o += 64;
  float* URC   = ws + o; o += 64;
  float* ROOT  = ws + o; o += 64;

  const float* XEP = XE + (size_t)N_LEAVES * HID;        // parent rows of xe

  // ---- weight re-layouts + embedding transpose ----
  transpose_E<<<(VOCAB * HID + 255) / 256, 256, 0, stream>>>(E_bu, ET);
  pack_wcat3<<<(64 * 192 + 255) / 256, 256, 0, stream>>>(W_z, W_r, W_h, BCAT);
  pack_uhB<<<16, 256, 0, stream>>>(U_h, BUH);
  pack_wqcat<<<(64 * 320 + 255) / 256, 256, 0, stream>>>(WQ, BWQ);
  pack_wkbd<<<(320 * 320 + 255) / 256, 256, 0, stream>>>(WK, BKBD);
  pack_wot<<<(320 * 64 + 255) / 256, 256, 0, stream>>>(WO, BWOT);
  prep_const<<<1, 64, 0, stream>>>(U_z, U_r, ln_b, UZC, URC);

  // ---- xe for all nodes ----
  xe_kernel<<<N_NODES, 64, 0, stream>>>(x_word, x_index, ET, XE);

  // ---- leaf GRU (batched via WMMA, 64x16 strips, B-fragment reuse) ----
  gemm_wmma_f16<<<dim3(N_LEAVES / 64, 192 / 16), 32, 0, stream>>>(XE, BCAT, G, N_LEAVES, 192, 64);
  leaf_zr<<<(N_LEAVES * HID + 255) / 256, 256, 0, stream>>>(G, UZC, URC, b_z, b_r, ln_b, ZST, T);
  gemm_wmma_f16<<<dim3(N_LEAVES / 64, 64 / 16), 32, 0, stream>>>(T, BUH, O2, N_LEAVES, 64, 64);
  leaf_final<<<(N_LEAVES * HID + 255) / 256, 256, 0, stream>>>(G, O2, ZST, b_h, ln_b, LEAFH);

  // ---- attention precompute (batched via WMMA) ----
  gemm_wmma_f16<<<dim3(N_PARENTS / 64, 320 / 16), 32, 0, stream>>>(XEP, BWQ, Q, N_PARENTS, 320, 64);
  gemm_wmma_f16<<<dim3(N_PARENTS / 64, 320 / 16), 32, 0, stream>>>(Q, BKBD, QW, N_PARENTS, 320, 320);
  gemm_wmma_f16<<<dim3(N_PARENTS / 64, 192 / 16), 32, 0, stream>>>(XEP, BCAT, GP, N_PARENTS, 192, 64);
  att_kernel<<<N_PARENTS, 64, 0, stream>>>(QW, XE, tree, LEAFH, ATT, PRE);
  gemm_wmma_f16<<<dim3(N_PARENTS / 64, 64 / 16), 32, 0, stream>>>(PRE, BWOT, VP, N_PARENTS, 64, 320);

  // ---- sequential chain (persistent workgroup, LDS-resident weights) ----
  size_t seq_lds = 34064 * sizeof(float);               // ~133 KB of the 320 KB WGP LDS
  seq_kernel<<<1, 256, seq_lds, stream>>>(BWOT, U_z, U_r, U_h, GP, VP, ATT,
                                          b_z, b_r, b_h, ln_g, ln_b, ROOT);

  // ---- loss ----
  loss_kernel<<<1, 64, 0, stream>>>(ROOT, W_out, b_out, y, out);

  (void)in_sizes; (void)n_in; (void)out_size; (void)ws_size;
}
